// MultiHeadAttention_2010044695245
// MI455X (gfx1250) — matrix operand order repaired and checked by reading, not compiled, not run on hardware
//
#include <hip/hip_runtime.h>

// Problem constants (match reference)
#define BB 2
#define SS 2048
#define DD 1024
#define HH 16
#define DKK 64

typedef __attribute__((ext_vector_type(16))) __bf16 v16bf;
typedef __attribute__((ext_vector_type(8)))  float  v8f;
typedef __attribute__((ext_vector_type(8)))  unsigned int v8u;

struct Frag32B { uint4 lo, hi; };

static __device__ __forceinline__ v16bf frag_cast(uint4 lo, uint4 hi) {
  Frag32B f{lo, hi};
  return __builtin_bit_cast(v16bf, f);
}
static __device__ __forceinline__ v16bf frag_cast8(v8u v) {
  return __builtin_bit_cast(v16bf, v);
}
// float -> bf16 via native conversion (RNE on gfx1250)
static __device__ __forceinline__ unsigned short f2bf(float f) {
  __bf16 h = (__bf16)f;
  return __builtin_bit_cast(unsigned short, h);
}
static __device__ __forceinline__ v8f wmma_bf16(v16bf a, v16bf b, v8f c) {
  // D = A(16x32 bf16) * B(32x16 bf16) + C(16x16 f32)
  return __builtin_amdgcn_wmma_f32_16x16x32_bf16(false, a, false, b, (short)0, c,
                                                 false, false);
}
static __device__ __forceinline__ void wait_dscnt0() {
#if __has_builtin(__builtin_amdgcn_s_wait_dscnt)
  __builtin_amdgcn_s_wait_dscnt(0);
#else
  asm volatile("s_wait_dscnt 0" ::: "memory");
#endif
}

// Convert 16 fp32 (A-fragment pattern: 16 contiguous floats)
// into a bf16 A fragment. Element j holds K = khalf*16 + j.
static __device__ __forceinline__ v16bf load_a_f32(const float* __restrict__ ap) {
  float4 a0 = *(const float4*)(ap);
  float4 a1 = *(const float4*)(ap + 4);
  float4 a2 = *(const float4*)(ap + 8);
  float4 a3 = *(const float4*)(ap + 12);
  v16bf a;
  a[0] = (__bf16)a0.x;  a[1] = (__bf16)a0.y;  a[2] = (__bf16)a0.z;  a[3] = (__bf16)a0.w;
  a[4] = (__bf16)a1.x;  a[5] = (__bf16)a1.y;  a[6] = (__bf16)a1.z;  a[7] = (__bf16)a1.w;
  a[8] = (__bf16)a2.x;  a[9] = (__bf16)a2.y;  a[10] = (__bf16)a2.z; a[11] = (__bf16)a2.w;
  a[12] = (__bf16)a3.x; a[13] = (__bf16)a3.y; a[14] = (__bf16)a3.z; a[15] = (__bf16)a3.w;
  return a;
}

// ---------------------------------------------------------------------------
// Kernel 1: transpose-convert W (fp32 [K][N] row-major) -> Wt (bf16 [N][K])
// so GEMM B-fragments (fixed column n, 16 consecutive K) are contiguous loads.
// ---------------------------------------------------------------------------
__global__ void __launch_bounds__(256)
wt_transpose_kernel(const float* __restrict__ Wq, const float* __restrict__ Wk,
                    const float* __restrict__ Wv, unsigned short* __restrict__ Wt) {
  __shared__ float tile[32][33];
  const int z = blockIdx.z;
  const float* W = (z == 0) ? Wq : ((z == 1) ? Wk : Wv);
  unsigned short* out = Wt + (size_t)z * DD * DD;
  const int nt = blockIdx.x * 32, kt = blockIdx.y * 32;
  const int tx = threadIdx.x, ty = threadIdx.y;
#pragma unroll
  for (int i = ty; i < 32; i += 8)
    tile[i][tx] = W[(size_t)(kt + i) * DD + nt + tx];
  __syncthreads();
#pragma unroll
  for (int i = ty; i < 32; i += 8)
    out[(size_t)(nt + i) * DD + kt + tx] = f2bf(tile[tx][i]);
}

// ---------------------------------------------------------------------------
// Kernel 2: fused QKV projection GEMM, M=B*S=4096, N=D=1024, K=D=1024.
// 128 threads = 4 waves; block tile 128(M)x64(N); each wave owns 32 rows x
// 64 cols -> 8 WMMAs per 32-deep k-step with A reused 4x and B reused 2x.
// W tile (64x32 bf16, 4 KB) is double-buffered in LDS: the next tile is
// staged into the idle buffer before the single per-iteration barrier.
// Q,K written as [B,H,S,DK] bf16; V written transposed [B,H,DK,S] bf16.
// ---------------------------------------------------------------------------
__global__ void __launch_bounds__(128)
qkv_proj_kernel(const float* __restrict__ qx, const float* __restrict__ kx,
                const float* __restrict__ vx, const float* __restrict__ bq,
                const float* __restrict__ bk, const float* __restrict__ bv,
                const unsigned short* __restrict__ Wt,
                unsigned short* __restrict__ Qh, unsigned short* __restrict__ Kh,
                unsigned short* __restrict__ Vt) {
  __shared__ v8u wlds[2][128];  // 2 buffers x (64 n-rows x 32 k bf16)

  const int z = blockIdx.z;
  const float* x = (z == 0) ? qx : ((z == 1) ? kx : vx);
  const float* bias = (z == 0) ? bq : ((z == 1) ? bk : bv);
  const unsigned short* w = Wt + (size_t)z * DD * DD;

  const int tid = threadIdx.x;
  const int lane = tid & 31, wid = tid >> 5;
  const int khalf = lane >> 4, l16 = lane & 15;
  const int rowbase = blockIdx.x * 128 + wid * 32;
  const int colbase = blockIdx.y * 64;
  const float* xrow0 = x + (size_t)(rowbase + l16) * DD;
  const float* xrow1 = xrow0 + (size_t)16 * DD;

  const int ldr = tid >> 1, ldh = tid & 1;  // cooperative W-tile staging
  const unsigned short* wsrc = w + (size_t)(colbase + ldr) * DD + ldh * 16;

  v8f acc[2][4] = {};

  // prologue: stage k=0 tile into buffer 0
  wlds[0][ldr * 2 + ldh] = *(const v8u*)(wsrc);
  __syncthreads();

  for (int k0 = 0; k0 < DD; k0 += 32) {
    const int p = (k0 >> 5) & 1;
    // stage next k-tile into the idle buffer (overlaps with compute below)
    if (k0 + 32 < DD)
      wlds[1 - p][ldr * 2 + ldh] = *(const v8u*)(wsrc + k0 + 32);

    v16bf a0 = load_a_f32(xrow0 + k0 + khalf * 16);
    v16bf a1 = load_a_f32(xrow1 + k0 + khalf * 16);
#pragma unroll
    for (int ns = 0; ns < 4; ++ns) {
      // B fragment: lane column n = ns*16+l16, K = khalf*16 + 0..15 contiguous
      v16bf bfrag = frag_cast8(wlds[p][(ns * 16 + l16) * 2 + khalf]);
      acc[0][ns] = wmma_bf16(a0, bfrag, acc[0][ns]);
      acc[1][ns] = wmma_bf16(a1, bfrag, acc[1][ns]);
    }
    __syncthreads();
  }

  // Epilogue: +bias, convert to bf16, scatter into head-major layouts.
#pragma unroll
  for (int rg = 0; rg < 2; ++rg) {
#pragma unroll
    for (int ns = 0; ns < 4; ++ns) {
      const int n = colbase + ns * 16 + l16;
      const float bz = bias[n];
      const int h = n >> 6, dk = n & 63;
#pragma unroll
      for (int r = 0; r < 8; ++r) {
        const int m = rowbase + rg * 16 + r + khalf * 8;  // C layout rows
        const int b = m >> 11;                            // S = 2048
        const int s = m & (SS - 1);
        const unsigned short bv16 = f2bf(acc[rg][ns][r] + bz);
        if (z == 2)
          Vt[((size_t)(b * HH + h) * DKK + dk) * SS + s] = bv16;
        else if (z == 0)
          Qh[((size_t)(b * HH + h) * SS + s) * DKK + dk] = bv16;
        else
          Kh[((size_t)(b * HH + h) * SS + s) * DKK + dk] = bv16;
      }
    }
  }
}

// ---------------------------------------------------------------------------
// Kernel 3: causal flash attention. One wave per 16-row Q tile; online
// softmax over 32-key blocks. scores: 2 WMMAs per 16x16 tile (DK=64=2xK32);
// P goes through per-wave LDS (C-layout -> A-layout transpose); PV: 4 WMMAs.
// ---------------------------------------------------------------------------
__global__ void __launch_bounds__(128)
flash_attn_kernel(const unsigned short* __restrict__ Qh,
                  const unsigned short* __restrict__ Kh,
                  const unsigned short* __restrict__ Vt,
                  float* __restrict__ out) {
  __shared__ __align__(16) unsigned short plds[4][16 * 32];

  const int tid = threadIdx.x;
  const int lane = tid & 31, wid = tid >> 5;
  const int khalf = lane >> 4, l16 = lane & 15;
  const int bh = blockIdx.y;
  const int b = bh >> 4, h = bh & (HH - 1);
  const int qtile = blockIdx.x * 4 + wid;
  const int qbase = qtile * 16;
  const unsigned short* Qp = Qh + (size_t)bh * SS * DKK;
  const unsigned short* Kp = Kh + (size_t)bh * SS * DKK;
  const unsigned short* Vp = Vt + (size_t)bh * DKK * SS;
  unsigned short* pl = plds[wid];

  // Q A-fragments, two dk chunks (0..31, 32..63); reused for every key block.
  v16bf qa[2];
  {
    const unsigned short* qrow = Qp + (size_t)(qbase + l16) * DKK;
#pragma unroll
    for (int c = 0; c < 2; ++c) {
      uint4 lo = *(const uint4*)(qrow + c * 32 + khalf * 16);
      uint4 hi = *(const uint4*)(qrow + c * 32 + khalf * 16 + 8);
      qa[c] = frag_cast(lo, hi);
    }
  }

  v8f o[4] = {};
  float mrow[8], lrow[8];
#pragma unroll
  for (int r = 0; r < 8; ++r) { mrow[r] = -1e30f; lrow[r] = 0.0f; }

  const int nkb = (qbase + 47) >> 5;  // causal: keys up to qbase+15 inclusive
  for (int kb = 0; kb < nkb; ++kb) {
    const int kbase = kb * 32;
    if (kb + 1 < nkb)
      __builtin_prefetch(Kp + (size_t)(kbase + 32) * DKK, 0, 1);

    // scores: two 16x16 f32 tiles over 32 keys
    v8f sc[2] = {};
#pragma unroll
    for (int t = 0; t < 2; ++t) {
      const unsigned short* krow = Kp + (size_t)(kbase + t * 16 + l16) * DKK;
#pragma unroll
      for (int c = 0; c < 2; ++c) {
        const unsigned short* p = krow + c * 32 + khalf * 16;
        uint4 lo = *(const uint4*)(p);
        uint4 hi = *(const uint4*)(p + 8);
        sc[t] = wmma_bf16(qa[c], frag_cast(lo, hi), sc[t]);
      }
    }

    // scale + causal mask + online softmax (rows live on 16-lane half-groups)
#pragma unroll
    for (int r = 0; r < 8; ++r) {
      const int row = qbase + r + khalf * 8;
      float s0 = sc[0][r] * 0.125f;  // 1/sqrt(64)
      float s1 = sc[1][r] * 0.125f;
      if (kbase + l16 > row) s0 = -1e30f;
      if (kbase + 16 + l16 > row) s1 = -1e30f;
      float t = fmaxf(s0, s1);
#pragma unroll
      for (int mm = 1; mm < 16; mm <<= 1) t = fmaxf(t, __shfl_xor(t, mm, 32));
      const float mnew = fmaxf(mrow[r], t);
      const float esc = __expf(mrow[r] - mnew);
      const float p0 = __expf(s0 - mnew);
      const float p1 = __expf(s1 - mnew);
      float rs = p0 + p1;
#pragma unroll
      for (int mm = 1; mm < 16; mm <<= 1) rs += __shfl_xor(rs, mm, 32);
      lrow[r] = lrow[r] * esc + rs;
      mrow[r] = mnew;
#pragma unroll
      for (int d = 0; d < 4; ++d) o[d][r] *= esc;
      // C-layout -> LDS row-major P tile [16 x 32] bf16
      pl[(r + khalf * 8) * 32 + l16] = f2bf(p0);
      pl[(r + khalf * 8) * 32 + 16 + l16] = f2bf(p1);
    }

    // LDS ops are in-order per wave on CDNA5; drain DScnt + block reordering.
    wait_dscnt0();

    // P A-fragment (row = l16, K = sk within block)
    uint4 plo = *(const uint4*)(pl + l16 * 32 + khalf * 16);
    uint4 phi = *(const uint4*)(pl + l16 * 32 + khalf * 16 + 8);
    v16bf pf = frag_cast(plo, phi);

    // PV: V^T rows are dk, so B fragment = 2 contiguous uint4 per lane
#pragma unroll
    for (int d = 0; d < 4; ++d) {
      const unsigned short* vrow =
          Vp + (size_t)(d * 16 + l16) * SS + kbase + khalf * 16;
      uint4 lo = *(const uint4*)(vrow);
      uint4 hi = *(const uint4*)(vrow + 8);
      o[d] = wmma_bf16(pf, frag_cast(lo, hi), o[d]);
    }
    asm volatile("" ::: "memory");  // keep next-iter P stores after these loads
  }

  // normalize and store fp32 output [B,S,D], d = h*64+dk
#pragma unroll
  for (int r = 0; r < 8; ++r) {
    const int s = qbase + r + khalf * 8;
    const float inv = 1.0f / lrow[r];
    float* orow = out + ((size_t)(b * SS + s)) * DD + h * DKK;
#pragma unroll
    for (int d = 0; d < 4; ++d) orow[d * 16 + l16] = o[d][r] * inv;
  }
}

// ---------------------------------------------------------------------------
extern "C" void kernel_launch(void* const* d_in, const int* in_sizes, int n_in,
                              void* d_out, int out_size, void* d_ws, size_t ws_size,
                              hipStream_t stream) {
  (void)in_sizes; (void)n_in; (void)out_size; (void)ws_size;
  const float* q  = (const float*)d_in[0];
  const float* k  = (const float*)d_in[1];
  const float* v  = (const float*)d_in[2];
  const float* Wq = (const float*)d_in[3];
  const float* bq = (const float*)d_in[4];
  const float* Wk = (const float*)d_in[5];
  const float* bk = (const float*)d_in[6];
  const float* Wv = (const float*)d_in[7];
  const float* bv = (const float*)d_in[8];
  // d_in[9] = causal mask, structurally known -> not read
  float* out = (float*)d_out;

  // workspace: Wt (3 x D*D bf16) | Qh | Kh | Vt (each B*S*D bf16) ~= 31.5 MB
  unsigned short* Wt = (unsigned short*)d_ws;
  unsigned short* Qh = Wt + (size_t)3 * DD * DD;
  unsigned short* Kh = Qh + (size_t)BB * SS * DD;
  unsigned short* Vt = Kh + (size_t)BB * SS * DD;

  wt_transpose_kernel<<<dim3(DD / 32, DD / 32, 3), dim3(32, 8), 0, stream>>>(
      Wq, Wk, Wv, Wt);
  qkv_proj_kernel<<<dim3((BB * SS) / 128, DD / 64, 3), dim3(128), 0, stream>>>(
      q, k, v, bq, bk, bv, Wt, Qh, Kh, Vt);
  flash_attn_kernel<<<dim3(SS / 64, BB * HH), dim3(128), 0, stream>>>(
      Qh, Kh, Vt, out);
}
